// NT_XentLoss_81003083202828
// MI455X (gfx1250) — matrix operand order repaired and checked by reading, not compiled, not run on hardware
//
#include <hip/hip_runtime.h>
#include <hip/hip_bf16.h>

typedef __attribute__((ext_vector_type(16))) _Float16 v16h;
typedef __attribute__((ext_vector_type(8)))  _Float16 v8h;
typedef __attribute__((ext_vector_type(8)))  float    v8f;

#define BB       4096
#define TWO_B    8192
#define DD       256
#define PANEL    128                 // columns staged in LDS per step
#define NPANEL   (TWO_B / PANEL)     // 64 panels total
#define MTILE    128                 // rows per workgroup (8 waves x 16)
#define CSPLIT   4                   // column strips (additive partial sums)
#define PPS      (NPANEL / CSPLIT)   // 16 panels per strip

__constant__ const float kInvT  = 1.0f / 0.07f;
__constant__ const float kLog2e = 1.4426950408889634f;

#if defined(__gfx1250__) && __has_builtin(__builtin_amdgcn_tensor_load_to_lds) && __has_builtin(__builtin_amdgcn_s_wait_tensorcnt)
#define USE_TDM 1
#else
#define USE_TDM 0
#endif

// Raw v_exp_f32: safe here, exponent argument is bounded in [-41.4, ~0].
#if __has_builtin(__builtin_amdgcn_exp2f)
#define EXP2F(x) __builtin_amdgcn_exp2f(x)
#else
#define EXP2F(x) exp2f(x)
#endif

// ---------------------------------------------------------------------------
// Kernel 1: L2-normalize rows of [z1;z2] and emit f16 zn[8192][256]
// ---------------------------------------------------------------------------
__global__ __launch_bounds__(256) void normalize_k(const float* __restrict__ z1,
                                                   const float* __restrict__ z2,
                                                   _Float16* __restrict__ zn) {
  const int w    = threadIdx.x >> 5;
  const int lane = threadIdx.x & 31;
  const int row  = blockIdx.x * 8 + w;                 // 0..8191, one wave per row
  const float* src = (row < BB) ? (z1 + (size_t)row * DD)
                                : (z2 + (size_t)(row - BB) * DD);
  float4 a = ((const float4*)src)[lane * 2 + 0];
  float4 b = ((const float4*)src)[lane * 2 + 1];
  float ss = a.x*a.x + a.y*a.y + a.z*a.z + a.w*a.w
           + b.x*b.x + b.y*b.y + b.z*b.z + b.w*b.w;
#pragma unroll
  for (int off = 16; off > 0; off >>= 1) ss += __shfl_xor(ss, off);
  const float sc = 1.0f / fmaxf(sqrtf(ss), 1e-12f);
  v8h o;
  o[0] = (_Float16)(a.x * sc); o[1] = (_Float16)(a.y * sc);
  o[2] = (_Float16)(a.z * sc); o[3] = (_Float16)(a.w * sc);
  o[4] = (_Float16)(b.x * sc); o[5] = (_Float16)(b.y * sc);
  o[6] = (_Float16)(b.z * sc); o[7] = (_Float16)(b.w * sc);
  *(v8h*)(zn + (size_t)row * DD + lane * 8) = o;
}

// ---------------------------------------------------------------------------
// Kernel 1b: target logits  tl_dot[i] = dot(zn[i], zn[i+B]),  i in [0, B)
// (sim[i, i+B] == sim[i+B, i] -> one dot serves both rows)
// ---------------------------------------------------------------------------
__global__ __launch_bounds__(256) void target_k(const _Float16* __restrict__ zn,
                                                float* __restrict__ tl_dot) {
  const int w    = threadIdx.x >> 5;
  const int lane = threadIdx.x & 31;
  const int i    = blockIdx.x * 8 + w;                 // 0..4095, one wave per pair
  const v8h va = *(const v8h*)(zn + (size_t)i * DD + lane * 8);
  const v8h vb = *(const v8h*)(zn + (size_t)(i + BB) * DD + lane * 8);
  float d = 0.0f;
#pragma unroll
  for (int j = 0; j < 8; ++j) d += (float)va[j] * (float)vb[j];
#pragma unroll
  for (int off = 16; off > 0; off >>= 1) d += __shfl_xor(d, off);
  if (lane == 0) tl_dot[i] = d;
}

// ---------------------------------------------------------------------------
// TDM: DMA one 128-row x 256-col f16 panel of zn into LDS (D# per ISA §8).
// 6-arg builtin: (u32x4 g0, i32x8 g1, i32x4 g2, i32x4 g3, i32x8 extra, i32 cpol)
// ---------------------------------------------------------------------------
#if USE_TDM
__device__ __forceinline__ void tdm_load_panel(const _Float16* gsrc, unsigned lds_off) {
  typedef unsigned int u32x4 __attribute__((ext_vector_type(4)));
  typedef int          i32x8 __attribute__((ext_vector_type(8)));
  typedef int          i32x4 __attribute__((ext_vector_type(4)));
  const unsigned long long ga = (unsigned long long)(size_t)gsrc;
  u32x4 g0;
  g0[0] = 1u;                                         // count=1, user descriptor
  g0[1] = lds_off;                                    // lds_addr (bytes)
  g0[2] = (unsigned)(ga & 0xFFFFFFFFu);               // global_addr[31:0]
  g0[3] = (unsigned)((ga >> 32) & 0x01FFFFFFu) | (2u << 30);   // addr[56:32], type=2
  i32x8 g1;
  g1[0] = (int)(1u << 16);          // workgroup_mask=0, data_size=1 (2 bytes)
  g1[1] = (int)((DD & 0xFFFFu) << 16);      // tensor_dim0[15:0] @ bits63:48
  g1[2] = (int)((TWO_B & 0xFFFFu) << 16);   // dim0 hi=0, tensor_dim1[15:0] @ bits95:80
  g1[3] = (int)((DD & 0xFFFFu) << 16);      // dim1 hi=0, tile_dim0=256 @ bits127:112
  g1[4] = (int)PANEL;               // tile_dim1=128, tile_dim2=0
  g1[5] = (int)DD;                  // tensor_dim0_stride = 256 elements
  g1[6] = 0;
  g1[7] = 0;
  i32x4 zz4 = {0, 0, 0, 0};         // groups 2/3 unused (2-D tensor)
  i32x8 zz8 = {0, 0, 0, 0, 0, 0, 0, 0};
  __builtin_amdgcn_tensor_load_to_lds(g0, g1, zz4, zz4, zz8, 0);
}
#endif

// ---------------------------------------------------------------------------
// Kernel 2: fused sim = zn*zn^T / T, partial sum-of-exp over one column strip.
// Fixed shift (rows unit-norm -> max logit == 1/T) makes strip sums additive,
// so grid = 64 row-blocks x CSPLIT col-strips with a deterministic final merge.
// ---------------------------------------------------------------------------
__global__ __launch_bounds__(256) void ntxent_main(const _Float16* __restrict__ zn,
                                                   float* __restrict__ partial) {
  __shared__ __align__(64) _Float16 sm[2][PANEL * DD];   // 2 x 64 KB double buffer
  const int tid  = threadIdx.x;
  const int w    = tid >> 5;
  const int lane = tid & 31;
  const int lg   = lane >> 4;      // half-wave group
  const int lm   = lane & 15;
  const int m0   = blockIdx.x * MTILE + w * 16;          // first row of this wave
  const int cb   = blockIdx.y;                           // column strip
  const int p0   = cb * PPS;                             // first panel of strip

  // Resident A fragments: 16 rows x 256 K (8 chunks of K=32), contiguous K order (as B):
  // lane group 0 holds K 0..15, group 1 holds K 16..31.
  const int rowA = m0 + lm;
  v16h a[8];
#pragma unroll
  for (int kc = 0; kc < 8; ++kc) {
    const _Float16* p = zn + (size_t)rowA * DD + kc * 32 + 16 * lg;
    v8h lo = *(const v8h*)p;
    v8h hi = *(const v8h*)(p + 8);
    a[kc] = __builtin_shufflevector(lo, hi, 0,1,2,3,4,5,6,7,8,9,10,11,12,13,14,15);
  }

  float s_[8];
#pragma unroll
  for (int r = 0; r < 8; ++r) s_[r] = 0.0f;
  const float A2 = kInvT * kLog2e;

#if USE_TDM
  const unsigned smbase = (unsigned)(size_t)(&sm[0][0]);
  if (tid < 32) tdm_load_panel(zn + (size_t)p0 * PANEL * DD, smbase);  // panel 0
#endif

  for (int pi = 0; pi < PPS; ++pi) {
#if USE_TDM
    const int buf = pi & 1;
    if (tid < 32) {
      if (pi + 1 < PPS) {
        tdm_load_panel(zn + (size_t)(p0 + pi + 1) * PANEL * DD,
                       smbase + (unsigned)((buf ^ 1) * PANEL * DD * 2));
        __builtin_amdgcn_s_wait_tensorcnt((short)1);     // panel pi landed
      } else {
        __builtin_amdgcn_s_wait_tensorcnt((short)0);
      }
    }
    __syncthreads();
#else
    const int buf = 0;
    {
      const float4* g4 = (const float4*)(zn + (size_t)(p0 + pi) * PANEL * DD);
      float4* d4 = (float4*)&sm[0][0];
      for (int i = tid; i < PANEL * DD / 8; i += 256) d4[i] = g4[i];
    }
    __syncthreads();
#endif
    const _Float16* smp = &sm[buf][0];

    for (int nt = 0; nt < PANEL / 16; ++nt) {
      v8f acc = {0.f, 0.f, 0.f, 0.f, 0.f, 0.f, 0.f, 0.f};
      // B layout: lane = column; group 0 holds K 0..15, group 1 K 16..31
      const _Float16* bp0 = smp + (size_t)(nt * 16 + lm) * DD + 16 * lg;
#pragma unroll
      for (int kc = 0; kc < 8; ++kc) {
        v16h bv = *(const v16h*)(bp0 + kc * 32);
        acc = __builtin_amdgcn_wmma_f32_16x16x32_f16(
            false, a[kc], false, bv, (short)0, acc, false, false);
      }
#pragma unroll
      for (int r = 0; r < 8; ++r)
        s_[r] += EXP2F(__builtin_fmaf(acc[r], A2, -A2));   // exp((d-1)/T)
    }
    __syncthreads();   // buffer `buf` may be overwritten next iteration
  }

  // Reduce across the 16 lanes sharing each row (stays inside half-wave).
#pragma unroll
  for (int r = 0; r < 8; ++r) {
#pragma unroll
    for (int off = 1; off < 16; off <<= 1) s_[r] += __shfl_xor(s_[r], off);
  }
  if (lm == 0) {
#pragma unroll
    for (int r = 0; r < 8; ++r) {
      const int gr = m0 + 8 * lg + r;
      partial[(size_t)cb * TWO_B + gr] = s_[r];
    }
  }
}

// ---------------------------------------------------------------------------
// Kernel 3: deterministic merge of strip partials + per-row loss + mean
//   loss_row = log(sum_c partial[c][row]) + (1 - tl_dot)/T
// ---------------------------------------------------------------------------
__global__ __launch_bounds__(256) void reduce_k(const float* __restrict__ partial,
                                                const float* __restrict__ tl_dot,
                                                float* __restrict__ out) {
  __shared__ float ss[8];
  float acc = 0.0f;
  for (int i = threadIdx.x; i < TWO_B; i += 256) {
    float s = 0.0f;
#pragma unroll
    for (int c = 0; c < CSPLIT; ++c) s += partial[(size_t)c * TWO_B + i];
    const float tl = tl_dot[(i < BB) ? i : i - BB];
    acc += logf(s) + (1.0f - tl) * kInvT;
  }
#pragma unroll
  for (int off = 16; off > 0; off >>= 1) acc += __shfl_xor(acc, off);
  if ((threadIdx.x & 31) == 0) ss[threadIdx.x >> 5] = acc;
  __syncthreads();
  if (threadIdx.x == 0) {
    float t = 0.0f;
#pragma unroll
    for (int i = 0; i < 8; ++i) t += ss[i];
    out[0] = t * (1.0f / (float)TWO_B);
  }
}

// ---------------------------------------------------------------------------
extern "C" void kernel_launch(void* const* d_in, const int* in_sizes, int n_in,
                              void* d_out, int out_size, void* d_ws, size_t ws_size,
                              hipStream_t stream) {
  const float* z1 = (const float*)d_in[0];
  const float* z2 = (const float*)d_in[1];
  float* out = (float*)d_out;

  _Float16* zn      = (_Float16*)d_ws;                                    // 4 MB
  float*    partial = (float*)((char*)d_ws + (size_t)TWO_B * DD * 2);     // 128 KB
  float*    tldot   = partial + (size_t)CSPLIT * TWO_B;                   // 16 KB

  normalize_k<<<TWO_B / 8, 256, 0, stream>>>(z1, z2, zn);
  target_k<<<BB / 8, 256, 0, stream>>>(zn, tldot);
  dim3 grid(TWO_B / MTILE, CSPLIT);
  ntxent_main<<<grid, 256, 0, stream>>>(zn, partial);
  reduce_k<<<1, 256, 0, stream>>>(partial, tldot, out);
}
